// UBlock_42288247996641
// MI455X (gfx1250) — compile-verified, matrix operand order checked by reading, not run on hardware
//
#include <hip/hip_runtime.h>
#include <hip/hip_bf16.h>
#include <cstddef>
#include <cstdint>

typedef __attribute__((ext_vector_type(16))) __bf16 v16bf;
typedef __attribute__((ext_vector_type(8)))  float  v8f;

#define BN_EPS 1e-4f

// ---- CDNA5 async global->LDS copy + counter fences (ISA 15.18.3 / 08_async_tensor) ----
__device__ __forceinline__ void async_g2l_b32(unsigned ldsOff, const void* gp) {
  asm volatile("global_load_async_to_lds_b32 %0, %1, off"
               :: "v"(ldsOff), "v"((unsigned long long)(uintptr_t)gp)
               : "memory");
}
__device__ __forceinline__ void wait_async0()  { asm volatile("s_wait_asynccnt 0x0"  ::: "memory"); }
__device__ __forceinline__ void wait_async16() { asm volatile("s_wait_asynccnt 0x10" ::: "memory"); }
__device__ __forceinline__ void wait_ds0()     { asm volatile("s_wait_dscnt 0x0"     ::: "memory"); }

// ---------------- elementwise / mask kernels ----------------

__global__ void k_mask_init(const int* __restrict__ mi, float* __restrict__ m, int n) {
  int i = blockIdx.x * blockDim.x + threadIdx.x;
  if (i < n) m[i] = mi[i] ? 1.0f : 0.0f;
}

__global__ void k_mask_mul(const float* __restrict__ x, const float* __restrict__ m,
                           float* __restrict__ y, int volShift, int n) {
  int i = blockIdx.x * blockDim.x + threadIdx.x;
  if (i < n) y[i] = x[i] * m[i & ((1 << volShift) - 1)];
}

__global__ void k_pool2(const float* __restrict__ mi, float* __restrict__ mo, int s /*log2 Do*/) {
  int i = blockIdx.x * blockDim.x + threadIdx.x;
  int Do = 1 << s;
  int volo = 1 << (3 * s);
  if (i >= volo) return;
  int w = i & (Do - 1), h = (i >> s) & (Do - 1), d = i >> (2 * s);
  int Di = 2 * Do;
  float v = 0.f;
  for (int kd = 0; kd < 2; ++kd)
    for (int kh = 0; kh < 2; ++kh)
      for (int kw = 0; kw < 2; ++kw) {
        int idx = ((2 * d + kd) * Di + (2 * h + kh)) * Di + (2 * w + kw);
        v = fmaxf(v, mi[idx]);
      }
  mo[i] = v;
}

// ---------------- masked batchnorm ----------------
__global__ void k_bn_stats(const float* __restrict__ x, const float* __restrict__ m,
                           float* __restrict__ mean, float* __restrict__ var, int vol) {
  __shared__ float ss[256], s2[256], sc[256];
  const int c = blockIdx.x;
  const float* xc = x + (size_t)c * vol;
  float a = 0.f, b = 0.f, cnt = 0.f;
  for (int i = threadIdx.x; i < vol; i += blockDim.x) {
    float mv = m[i];
    float xm = xc[i] * mv;   // mask is 0/1 -> x*x*m == x*(x*m)
    a += xm;
    b += xc[i] * xm;
    cnt += mv;
  }
  int t = threadIdx.x;
  ss[t] = a; s2[t] = b; sc[t] = cnt;
  __syncthreads();
  for (int s = 128; s > 0; s >>= 1) {
    if (t < s) { ss[t] += ss[t + s]; s2[t] += s2[t + s]; sc[t] += sc[t + s]; }
    __syncthreads();
  }
  if (t == 0) {
    float cN = fmaxf(sc[0], 1.0f);
    float mu = ss[0] / cN;
    mean[c] = mu;
    var[c]  = s2[0] / cN - mu * mu;
  }
}

__global__ void k_bn_relu(const float* __restrict__ x, const float* __restrict__ m,
                          const float* __restrict__ mean, const float* __restrict__ var,
                          const float* __restrict__ g, const float* __restrict__ b,
                          float* __restrict__ y, int volShift, int n) {
  int i = blockIdx.x * blockDim.x + threadIdx.x;
  if (i >= n) return;
  int c = i >> volShift;
  int sp = i & ((1 << volShift) - 1);
  float ys = (x[i] - mean[c]) * rsqrtf(var[c] + BN_EPS) * g[c] + b[c];
  y[i] = fmaxf(ys, 0.f) * m[sp];
}

// ---------------- transposed conv k=2 s=2 (SparseInverseConv3d) ----------------
// Weight layout IODHW: w[ci][co][kd][kh][kw]; one input voxel feeds each output voxel.
__global__ void k_upconv(const float* __restrict__ x, const float* __restrict__ w,
                         const float* __restrict__ m, float* __restrict__ y,
                         int Cin, int Cout, int s /*log2 Do*/) {
  int v = blockIdx.x * blockDim.x + threadIdx.x;
  int Do = 1 << s;
  int volo = 1 << (3 * s);
  if (v >= volo) return;
  int co = blockIdx.y;
  int wp = v & (Do - 1), hp = (v >> s) & (Do - 1), dp = v >> (2 * s);
  int Di = Do >> 1;
  int voli = volo >> 3;
  int vi = (((dp >> 1) * Di) + (hp >> 1)) * Di + (wp >> 1);
  int tap = ((dp & 1) << 2) | ((hp & 1) << 1) | (wp & 1);
  float sum = 0.f;
  for (int ci = 0; ci < Cin; ++ci)
    sum += x[(size_t)ci * voli + vi] * w[((size_t)ci * Cout + co) * 8 + tap];
  y[(size_t)co * volo + v] = sum * m[v];
}

// ---------------- implicit-GEMM conv via v_wmma_f32_16x16x32_bf16 ----------------
// One wave per block; tile = 16 output channels (M) x 16 consecutive-W voxels (N).
// Software-pipelined, double-buffered LDS staging:
//   - B tile: GLOBAL_LOAD_ASYNC_TO_LDS_B32 for interior taps; next chunk's asyncs are
//     issued before waiting, then s_wait_asynccnt 0x10 drains only the older 16
//     (async loads complete in order) so global latency overlaps the WMMA.
//   - A tile: LDS stores; single-wave workgroup -> explicit s_wait_dscnt 0x0 gives
//     cross-lane visibility (same-wave DS ops are in order), no barrier needed.
template <int KS, int STRIDE>
__global__ __launch_bounds__(32) void k_conv_wmma(
    const float* __restrict__ x, const float* __restrict__ w,
    const float* __restrict__ mask, float* __restrict__ y,
    int Cin, int D, int H, int W, int Dout, int Hout, int Wout,
    int wshift, int accf) {
  constexpr int PAD = (STRIDE == 1) ? (KS - 1) / 2 : 0;
  const int lane = threadIdx.x;
  const int row  = lane & 15;   // M for A, N for B/C/D
  const int grp  = lane >> 4;   // lane-group: selects K sub-range / M-half

  const int ws = blockIdx.x & ((1u << wshift) - 1);
  const int co0 = (blockIdx.x >> wshift) * 16;
  const int h = blockIdx.y;
  const int d = blockIdx.z;

  __shared__ float Asf[2][16][32];  // [buf][co][ci_local]
  __shared__ float Bt[2][16][32];   // [buf][voxel][ci_local]

  v8f acc = {};
  const size_t volin = (size_t)D * H * W;
  const int nChunk = Cin >> 5;           // 32-channel chunks (1,2,3,4)
  const int NIT = KS * KS * KS * nChunk;

  struct Idx { int kd, kh, kw, cc; };
  auto adv = [&](Idx t) {
    if (++t.cc == nChunk) { t.cc = 0;
      if (++t.kw == KS) { t.kw = 0;
        if (++t.kh == KS) { t.kh = 0; ++t.kd; } } }
    return t;
  };

  // Stage one (tap, chunk) tile into LDS buffer `buf`; returns true if async was used.
  auto stage = [&](const Idx& I, int buf) -> bool {
    const int din = d * STRIDE - PAD + I.kd;
    const int hin = h * STRIDE - PAD + I.kh;
    const int wbase = (ws * 16) * STRIDE - PAD + I.kw;
    const int c0 = I.cc << 5;
    const bool rowOK = (din >= 0) && (din < D) && (hin >= 0) && (hin < H);
    const bool interior = rowOK && (wbase >= 0) && (wbase + 15 * STRIDE < W);
    {  // A: lane -> output channel co0+row, 16 input channels (strided gather)
      const int co = co0 + row;
#pragma unroll
      for (int j = 0; j < 16; ++j) {
        const int cil = grp * 16 + j;
        const size_t wi =
            ((((size_t)co * Cin + (c0 + cil)) * KS + I.kd) * KS + I.kh) * KS + I.kw;
        Asf[buf][row][cil] = w[wi];
      }
    }
    // B: channel c0+lane; column t for all 32 channels per async instruction
    const float* xrow = x + (size_t)(c0 + lane) * volin + (size_t)(din * H + hin) * W;
    if (interior) {
#pragma unroll
      for (int t = 0; t < 16; ++t)
        async_g2l_b32((unsigned)(uintptr_t)&Bt[buf][t][lane], xrow + wbase + t * STRIDE);
    } else {
#pragma unroll
      for (int t = 0; t < 16; ++t) {
        const int win = wbase + t * STRIDE;
        float v = 0.f;
        if (rowOK && win >= 0 && win < W) v = xrow[win];
        Bt[buf][t][lane] = v;
      }
    }
    return interior;
  };

  Idx cur{0, 0, 0, 0};
  (void)stage(cur, 0);          // prologue: stage chunk 0 into buffer 0
  Idx nxt = adv(cur);
  int curBuf = 0;

  for (int iter = 0; iter < NIT; ++iter) {
    const bool hasNext = (iter + 1) < NIT;
    bool nextAsync = false;
    wait_ds0();                 // drain prior frag reads + current buf's DS staging
    if (hasNext) nextAsync = stage(nxt, curBuf ^ 1);
    if (hasNext && nextAsync) wait_async16();   // only drain the OLDER 16 asyncs
    else                      wait_async0();    // boundary/last: drain everything
    wait_ds0();                 // A stores of next iter don't matter; cur's are drained

    // Fragment build per ISA 7.12.2 wave32 layouts: lane reads two contiguous
    // 8-float runs (ci = grp*8.. and 16+grp*8..) -> float4 LDS reads.
    const float4* ap = reinterpret_cast<const float4*>(&Asf[curBuf][row][0]);
    const float4* bp = reinterpret_cast<const float4*>(&Bt[curBuf][row][0]);
    float4 a0 = ap[grp * 2], a1 = ap[grp * 2 + 1];
    float4 a2 = ap[4 + grp * 2], a3 = ap[5 + grp * 2];
    float4 b0 = bp[grp * 2], b1 = bp[grp * 2 + 1];
    float4 b2 = bp[4 + grp * 2], b3 = bp[5 + grp * 2];
    v16bf av, bv;
    av[0] = (__bf16)a0.x; av[1] = (__bf16)a0.y; av[2] = (__bf16)a0.z; av[3] = (__bf16)a0.w;
    av[4] = (__bf16)a1.x; av[5] = (__bf16)a1.y; av[6] = (__bf16)a1.z; av[7] = (__bf16)a1.w;
    av[8] = (__bf16)a2.x; av[9] = (__bf16)a2.y; av[10] = (__bf16)a2.z; av[11] = (__bf16)a2.w;
    av[12] = (__bf16)a3.x; av[13] = (__bf16)a3.y; av[14] = (__bf16)a3.z; av[15] = (__bf16)a3.w;
    bv[0] = (__bf16)b0.x; bv[1] = (__bf16)b0.y; bv[2] = (__bf16)b0.z; bv[3] = (__bf16)b0.w;
    bv[4] = (__bf16)b1.x; bv[5] = (__bf16)b1.y; bv[6] = (__bf16)b1.z; bv[7] = (__bf16)b1.w;
    bv[8] = (__bf16)b2.x; bv[9] = (__bf16)b2.y; bv[10] = (__bf16)b2.z; bv[11] = (__bf16)b2.w;
    bv[12] = (__bf16)b3.x; bv[13] = (__bf16)b3.y; bv[14] = (__bf16)b3.z; bv[15] = (__bf16)b3.w;

    acc = __builtin_amdgcn_wmma_f32_16x16x32_bf16(
        false, av, false, bv, (short)0, acc, false, false);

    nxt = adv(nxt);
    curBuf ^= 1;
  }

  // C/D layout: lane holds N=row; VGPR r holds M = r + 8*grp.
  const size_t volo = (size_t)Dout * Hout * Wout;
  const size_t sp = (size_t)(d * Hout + h) * Wout + ws * 16 + row;
  const float mval = mask[sp];
#pragma unroll
  for (int r = 0; r < 8; ++r) {
    const int M = r + 8 * grp;
    const size_t oi = (size_t)(co0 + M) * volo + sp;
    float prev = accf ? y[oi] : 0.f;
    y[oi] = prev + mval * acc[r];
  }
}

// ---------------- host orchestration ----------------

extern "C" void kernel_launch(void* const* d_in, const int* in_sizes, int n_in,
                              void* d_out, int out_size, void* d_ws, size_t ws_size,
                              hipStream_t stream) {
  (void)in_sizes; (void)n_in; (void)out_size; (void)ws_size;
  const int vol0 = 64 * 64 * 64, vol1 = 32 * 32 * 32, vol2 = 16 * 16 * 16;

  float* WS = (float*)d_ws;
  float* MEAN = WS;                   // 128
  float* VAR  = WS + 128;             // 128
  float* M0 = WS + 256;               // vol0
  float* M1 = M0 + vol0;              // vol1
  float* M2 = M1 + vol1;              // vol2
  float* A0  = M2 + vol2;             // 32*vol0
  float* B0  = A0 + 32 * vol0;        // 32*vol0
  float* C0  = B0 + 32 * vol0;        // 32*vol0
  float* ID0 = C0 + 32 * vol0;        // 32*vol0  (CAT0 = [ID0 | UP0], contiguous 64ch)
  float* UP0 = ID0 + 32 * vol0;       // 32*vol0
  float* A1  = UP0 + 32 * vol0;       // 64*vol1
  float* B1  = A1 + 64 * vol1;        // 64*vol1
  float* C1  = B1 + 64 * vol1;        // 64*vol1
  float* ID1 = C1 + 64 * vol1;        // 64*vol1  (CAT1 = [ID1 | UP1], contiguous 128ch)
  float* UP1 = ID1 + 64 * vol1;       // 64*vol1
  float* CT1 = UP1 + 64 * vol1;       // 128*vol1 bn temp
  float* A2  = CT1 + 128 * vol1;      // 96*vol2
  float* B2  = A2 + 96 * vol2;        // 96*vol2
  float* C2  = B2 + 96 * vol2;        // 96*vol2  -> total ~58.1M floats (~233 MB)
  float* CT0 = A0;                    // alias: 64*vol0 bn temp over [A0|B0] (both free then)

  // ---- unpack inputs in setup_inputs() flattening order ----
  int idx = 0;
  auto nf = [&]() { return (const float*)d_in[idx++]; };
  struct RP { const float *g1, *b1, *w1, *g2, *b2, *w2, *wi; };
  auto res = [&](bool haswi) {
    RP p; p.g1 = nf(); p.b1 = nf(); p.w1 = nf();
    p.g2 = nf(); p.b2 = nf(); p.w2 = nf();
    p.wi = haswi ? nf() : nullptr; return p;
  };
  struct DP { const float *g, *b, *w; };
  auto dpp = [&]() { DP p; p.g = nf(); p.b = nf(); p.w = nf(); return p; };

  const float* x_in = nf();
  RP enc0[2] = {res(false), res(false)};
  DP down0 = dpp();
  RP enc1[2] = {res(false), res(false)};
  DP down1 = dpp();
  RP mid[2] = {res(false), res(false)};
  DP dec1 = dpp();
  RP tail1[2] = {res(true), res(false)};
  DP dec0 = dpp();
  RP tail0[2] = {res(true), res(false)};
  const int* mask_in = (const int*)d_in[idx++];

  // ---- launch helpers ----
  auto log2i = [](int v) { int s = 0; while ((1 << s) < v) ++s; return s; };
  auto bn = [&](const float* xin, const float* m, const float* g, const float* b,
                float* yout, int C, int vol) {
    k_bn_stats<<<C, 256, 0, stream>>>(xin, m, MEAN, VAR, vol);
    int n = C * vol;
    k_bn_relu<<<(n + 255) / 256, 256, 0, stream>>>(xin, m, MEAN, VAR, g, b, yout,
                                                   log2i(vol), n);
  };
  auto conv3 = [&](const float* xin, const float* wp, const float* m, float* yout,
                   int Cin, int Cout, int Dm, int accf) {
    int wsegs = Dm / 16;
    dim3 grid(wsegs * (Cout / 16), Dm, Dm);
    k_conv_wmma<3, 1><<<grid, 32, 0, stream>>>(xin, wp, m, yout, Cin, Dm, Dm, Dm,
                                               Dm, Dm, Dm, log2i(wsegs), accf);
  };
  auto conv1 = [&](const float* xin, const float* wp, const float* m, float* yout,
                   int Cin, int Cout, int Dm, int accf) {
    int wsegs = Dm / 16;
    dim3 grid(wsegs * (Cout / 16), Dm, Dm);
    k_conv_wmma<1, 1><<<grid, 32, 0, stream>>>(xin, wp, m, yout, Cin, Dm, Dm, Dm,
                                               Dm, Dm, Dm, log2i(wsegs), accf);
  };
  auto conv2 = [&](const float* xin, const float* wp, const float* m, float* yout,
                   int Cin, int Cout, int Din) {
    int Dou = Din / 2;
    int wsegs = Dou / 16;
    dim3 grid(wsegs * (Cout / 16), Dou, Dou);
    k_conv_wmma<2, 2><<<grid, 32, 0, stream>>>(xin, wp, m, yout, Cin, Din, Din, Din,
                                               Dou, Dou, Dou, log2i(wsegs), 0);
  };
  auto resblock = [&](float* X, float* T1, float* T2, const float* m,
                      const RP& p, int C, int Dm) {
    // Cin==Cout, identity skip: conv(w2) accumulates into X.
    bn(X, m, p.g1, p.b1, T1, C, Dm * Dm * Dm);
    conv3(T1, p.w1, m, T2, C, C, Dm, 0);
    bn(T2, m, p.g2, p.b2, T1, C, Dm * Dm * Dm);
    conv3(T1, p.w2, m, X, C, C, Dm, 1);
  };

  // ---- masks ----
  k_mask_init<<<(vol0 + 255) / 256, 256, 0, stream>>>(mask_in, M0, vol0);
  k_pool2<<<(vol1 + 255) / 256, 256, 0, stream>>>(M0, M1, 5);
  k_pool2<<<(vol2 + 255) / 256, 256, 0, stream>>>(M1, M2, 4);

  // ---- level 0 encoder ----
  k_mask_mul<<<(32 * vol0 + 255) / 256, 256, 0, stream>>>(x_in, M0, A0, 18, 32 * vol0);
  for (int r = 0; r < 2; ++r) resblock(A0, B0, C0, M0, enc0[r], 32, 64);
  hipMemcpyAsync(ID0, A0, (size_t)32 * vol0 * sizeof(float), hipMemcpyDeviceToDevice, stream);

  // ---- down 0 ----
  bn(A0, M0, down0.g, down0.b, B0, 32, vol0);
  conv2(B0, down0.w, M1, A1, 32, 64, 64);

  // ---- level 1 encoder ----
  for (int r = 0; r < 2; ++r) resblock(A1, B1, C1, M1, enc1[r], 64, 32);
  hipMemcpyAsync(ID1, A1, (size_t)64 * vol1 * sizeof(float), hipMemcpyDeviceToDevice, stream);

  // ---- down 1 ----
  bn(A1, M1, down1.g, down1.b, B1, 64, vol1);
  conv2(B1, down1.w, M2, A2, 64, 96, 32);

  // ---- mid ----
  for (int r = 0; r < 2; ++r) resblock(A2, B2, C2, M2, mid[r], 96, 16);

  // ---- decode 2 -> 1 : up + concat(ID1, up) ----
  bn(A2, M2, dec1.g, dec1.b, B2, 96, vol2);
  {
    dim3 g((vol1 + 255) / 256, 64);
    k_upconv<<<g, 256, 0, stream>>>(B2, dec1.w, M1, UP1, 96, 64, 5);
  }
  {  // tail1[0]: res_p(128, 64) on CAT1 = ID1 (128 channels contiguous)
    const RP& p = tail1[0];
    bn(ID1, M1, p.g1, p.b1, CT1, 128, vol1);
    conv3(CT1, p.w1, M1, A1, 128, 64, 32, 0);
    bn(A1, M1, p.g2, p.b2, B1, 64, vol1);
    conv3(B1, p.w2, M1, C1, 64, 64, 32, 0);
    conv1(ID1, p.wi, M1, C1, 128, 64, 32, 1);  // identity path, accumulated
  }
  resblock(C1, B1, A1, M1, tail1[1], 64, 32);

  // ---- decode 1 -> 0 : up + concat(ID0, up) ----
  bn(C1, M1, dec0.g, dec0.b, B1, 64, vol1);
  {
    dim3 g((vol0 + 255) / 256, 32);
    k_upconv<<<g, 256, 0, stream>>>(B1, dec0.w, M0, UP0, 64, 32, 6);
  }
  {  // tail0[0]: res_p(64, 32) on CAT0 = ID0 (64 channels contiguous)
    const RP& p = tail0[0];
    bn(ID0, M0, p.g1, p.b1, CT0, 64, vol0);     // CT0 aliases [A0|B0] (both free here)
    conv3(CT0, p.w1, M0, C0, 64, 32, 64, 0);
    bn(C0, M0, p.g2, p.b2, A0, 32, vol0);       // CT0 dead past this point
    conv3(A0, p.w2, M0, B0, 32, 32, 64, 0);
    conv1(ID0, p.wi, M0, B0, 64, 32, 64, 1);    // identity path, accumulated
  }
  {  // tail0[1]: res_p(32, 32) on B0, final result into d_out
    const RP& p = tail0[1];
    bn(B0, M0, p.g1, p.b1, A0, 32, vol0);
    conv3(A0, p.w1, M0, C0, 32, 32, 64, 0);
    bn(C0, M0, p.g2, p.b2, A0, 32, vol0);
    hipMemcpyAsync(d_out, B0, (size_t)32 * vol0 * sizeof(float), hipMemcpyDeviceToDevice, stream);
    conv3(A0, p.w2, M0, (float*)d_out, 32, 32, 64, 1);
  }
}
